// CRF_71571335021248
// MI455X (gfx1250) — compile-verified, matrix operand order checked by reading, not run on hardware
//
#include <hip/hip_runtime.h>

typedef _Float16 half_t;
typedef __attribute__((ext_vector_type(16))) _Float16 v16h;
typedef __attribute__((ext_vector_type(8)))  _Float16 v8h;
typedef __attribute__((ext_vector_type(8)))  float    v8f;
typedef __attribute__((ext_vector_type(4)))  float    v4f;

#define Bc 128
#define Tc 1024
#define Lc 96
#define EST 20   // ebuf row stride in floats (padded: dodges bank conflicts)

// Load a v16h WMMA operand fragment from a row-major [*, 96] f16 LDS array.
// Per ISA 7.12.2 (16-bit operands, 16x32): halfs 0..7 hold K = kb..kb+7,
// halfs 8..15 hold K = kb+16..kb+23 (caller folds the lane-half +8 K offset
// into kb). Both 16B chunks are aligned (row stride 192B, kb multiple of 8).
__device__ __forceinline__ v16h ld_frag(const half_t* base, int row, int kb) {
  const half_t* rp = base + row * Lc + kb;
  v8h lo = *(const v8h*)(rp);
  v8h hi = *(const v8h*)(rp + 16);
  v16h o;
#pragma unroll
  for (int i = 0; i < 8; ++i) { o[i] = lo[i]; o[i + 8] = hi[i]; }
  return o;
}

// Row-sum of this wave's 16x16 C-tile into rows[0..15] (LDS atomics).
// Stage row-major in per-wave LDS scratch (8 ds_store_b32), re-read one
// row-half per lane (2 ds_load_b128), 7 VALU adds, 1 ds_add_f32.
__device__ __forceinline__ void rowreduce(const float ev[8], float* ebufw,
                                          float* rows, int hi, int col16) {
#pragma unroll
  for (int r = 0; r < 8; ++r) ebufw[(8 * hi + r) * EST + col16] = ev[r];
  const v4f* rp = (const v4f*)(ebufw + col16 * EST + 8 * hi);
  v4f a = rp[0], b = rp[1];
  float s = ((a.x + a.y) + (a.z + a.w)) + ((b.x + b.y) + (b.z + b.w));
  atomicAdd(&rows[col16], s);   // both halves contribute their half-row sum
}

// ---------------------------------------------------------------------------
// Kernel 1: point_score + trans_score per batch element -> d_ws[b]
// ---------------------------------------------------------------------------
__global__ void crf_scores_kernel(const float* __restrict__ inp,
                                  const int* __restrict__ lab,
                                  const float* __restrict__ trans,
                                  float* __restrict__ scores) {
  const int b = blockIdx.x;
  const int tid = threadIdx.x;
  const int* lb = lab + (size_t)b * Tc;
  const float* xb = inp + (size_t)b * Tc * Lc;

  float acc = 0.f;
  for (int t = tid; t < Tc; t += 256) {
    int l0 = lb[t];
    acc += xb[(size_t)t * Lc + l0];
    if (t + 1 < Tc) acc += trans[l0 * Lc + lb[t + 1]];
  }
#pragma unroll
  for (int m = 16; m >= 1; m >>= 1) acc += __shfl_xor(acc, m, 32);

  __shared__ float red[8];
  if ((tid & 31) == 0) red[tid >> 5] = acc;
  __syncthreads();
  if (tid == 0) {
    float s = 0.f;
#pragma unroll
    for (int i = 0; i < 8; ++i) s += red[i];
    scores[b] = s;
  }
}

// ---------------------------------------------------------------------------
// Kernel 2: forward scan. One block = 16 batch rows, 6 waves = 6 N-tiles.
// Deferred-normalization recurrence (one barrier per step):
//   q_t = (q_{t-1} @ exp(trans)) * exp(x_t) * rcp(R_{t-1})   (bounded, f16-safe)
//   R_t = row-sum(q_t) ;  C += log(R_t)  =>  log_norm = C after last step
// pmat double-buffered, rowsum triple-buffered; time loop unrolled by 6
// (= lcm(2,3)) so all buffer indices are compile-time constants.
// ---------------------------------------------------------------------------
__launch_bounds__(192, 1)
__global__ void crf_forward_kernel(const float* __restrict__ inp,
                                   const float* __restrict__ trans,
                                   const float* __restrict__ scores,
                                   float* __restrict__ out) {
  __shared__ __align__(16) half_t expTt[Lc * Lc];   // exp(trans)^T : [n][k]
  __shared__ __align__(16) half_t pbuf[2][16 * Lc]; // q : [batch-row][l]
  __shared__ __align__(16) float  ebuf[6 * 16 * EST];
  __shared__ __align__(16) float  rowsum[3][16];

  const int tid   = threadIdx.x;
  const int w     = tid >> 5;        // wave id 0..5 -> N-tile
  const int lane  = tid & 31;
  const int hi    = lane >> 4;       // lane half: K offset +8, C rows +8
  const int col16 = lane & 15;
  const int col   = w * 16 + col16;  // global label column of this lane
  const int b0    = blockIdx.x * 16;
  float* ebufw = ebuf + w * 16 * EST;

  for (int i = tid; i < Lc * Lc; i += 192) {
    int k = i / Lc, n = i - k * Lc;
    expTt[n * Lc + k] = (half_t)__expf(trans[k * Lc + n]);
  }
  if (tid < 16) { rowsum[0][tid] = 0.f; rowsum[1][tid] = 0.f; rowsum[2][tid] = 0.f; }
  __syncthreads();

  // Loop-invariant B fragments in registers for all 1023 steps.
  v16h bf0 = ld_frag(expTt, col, 0 + hi * 8);
  v16h bf1 = ld_frag(expTt, col, 32 + hi * 8);
  v16h bf2 = ld_frag(expTt, col, 64 + hi * 8);

  // Per-lane global base: element r covers (batch row b0 + r + 8*hi, col).
  const float* xp = inp + ((size_t)(b0 + 8 * hi) * Tc) * Lc + col;
  const size_t rstride = (size_t)Tc * Lc;

  // ---- t = 0: q0 = exp(x0) (unnormalized) ----
  float xcur[8], ev[8];
#pragma unroll
  for (int r = 0; r < 8; ++r) xcur[r] = xp[r * rstride];
#pragma unroll
  for (int r = 0; r < 8; ++r) {
    ev[r] = __expf(xcur[r]);
    pbuf[0][(8 * hi + r) * Lc + col] = (half_t)ev[r];
  }
  rowreduce(ev, ebufw, rowsum[0], hi, col16);
  __syncthreads();

  // Prime x(t=1).
#pragma unroll
  for (int r = 0; r < 8; ++r) xcur[r] = xp[r * rstride + Lc];

  float Creg = 0.f;

  // One scan step. rd/wr: pmat parities; q0/q1/q2: rowsum prev/cur/to-zero.
  auto crf_step = [&](int t, int rd, int wr, int q0, int q1, int q2) {
    // Prefetch x(t+1) early to overlap with WMMA + transcendentals.
    const int tn = (t + 1 < Tc) ? (t + 1) : t;
    float xnext[8];
#pragma unroll
    for (int r = 0; r < 8; ++r) xnext[r] = xp[r * rstride + (size_t)tn * Lc];

    // rcp of previous row sums: off the WMMA critical path (data is ready
    // since the last barrier). 2 ds_load_b128 + 8 v_rcp_f32.
    const v4f* rsp = (const v4f*)(&rowsum[q0][8 * hi]);
    v4f Ra = rsp[0], Rb = rsp[1];
    float Ri[8];
    Ri[0] = __builtin_amdgcn_rcpf(Ra.x); Ri[1] = __builtin_amdgcn_rcpf(Ra.y);
    Ri[2] = __builtin_amdgcn_rcpf(Ra.z); Ri[3] = __builtin_amdgcn_rcpf(Ra.w);
    Ri[4] = __builtin_amdgcn_rcpf(Rb.x); Ri[5] = __builtin_amdgcn_rcpf(Rb.y);
    Ri[6] = __builtin_amdgcn_rcpf(Rb.z); Ri[7] = __builtin_amdgcn_rcpf(Rb.w);

    // acc = q_{t-1} @ exp(trans)  (16x16 tile, K = 96 via 3 x K32 f16 WMMAs)
    const half_t* pm = pbuf[rd];
    v16h a0 = ld_frag(pm, col16, 0 + hi * 8);
    v16h a1 = ld_frag(pm, col16, 32 + hi * 8);
    v16h a2 = ld_frag(pm, col16, 64 + hi * 8);
    v8f c = {};
    c = __builtin_amdgcn_wmma_f32_16x16x32_f16(false, a0, false, bf0, (short)0, c, false, false);
    c = __builtin_amdgcn_wmma_f32_16x16x32_f16(false, a1, false, bf1, (short)0, c, false, false);
    c = __builtin_amdgcn_wmma_f32_16x16x32_f16(false, a2, false, bf2, (short)0, c, false, false);

    // q_t = acc * exp(x_t) * rcp(R_{t-1})
#pragma unroll
    for (int r = 0; r < 8; ++r) {
      float e = c[r] * __expf(xcur[r]) * Ri[r];
      ev[r] = e;
      pbuf[wr][(8 * hi + r) * Lc + col] = (half_t)e;
    }
    rowreduce(ev, ebufw, rowsum[q1], hi, col16);

    if (tid < 16) {
      Creg += __logf(rowsum[q0][tid]);  // consume R_{t-1}
      rowsum[q2][tid] = 0.f;            // ready the t+1 accumulation buffer
    }
    __syncthreads();

#pragma unroll
    for (int r = 0; r < 8; ++r) xcur[r] = xnext[r];
  };

  // 1023 steps: 170 blocks of 6 (all buffer indices constant) + 3 epilogue.
  int tb = 1;
#pragma unroll 1
  for (; tb + 5 < Tc; tb += 6) {
#pragma unroll
    for (int u = 0; u < 6; ++u)
      crf_step(tb + u, u & 1, (u + 1) & 1, u % 3, (u + 1) % 3, (u + 2) % 3);
  }
#pragma unroll
  for (int u = 0; u < 3; ++u)
    crf_step(tb + u, u & 1, (u + 1) & 1, u % 3, (u + 1) % 3, (u + 2) % 3);

  // log_norm = C + log(R_{1023})  (1023 % 3 == 0 -> rowsum[0]); minus gold.
  if (tid < 16) {
    Creg += __logf(rowsum[0][tid]);
    out[b0 + tid] = Creg - scores[b0 + tid];
  }
}

// ---------------------------------------------------------------------------
extern "C" void kernel_launch(void* const* d_in, const int* in_sizes, int n_in,
                              void* d_out, int out_size, void* d_ws, size_t ws_size,
                              hipStream_t stream) {
  const float* inp   = (const float*)d_in[0];   // (B, T, L) f32
  const int*   lab   = (const int*)d_in[1];     // (B, T) i32
  const float* trans = (const float*)d_in[2];   // (L, L) f32
  float* out    = (float*)d_out;                // (B, 1) f32
  float* scores = (float*)d_ws;                 // B floats of scratch

  crf_scores_kernel<<<Bc, 256, 0, stream>>>(inp, lab, trans, scores);
  crf_forward_kernel<<<Bc / 16, 192, 0, stream>>>(inp, trans, scores, out);
}
